// EfficientMemoryGEMM_88390426952067
// MI455X (gfx1250) — matrix operand order repaired and checked by reading, not compiled, not run on hardware
//
#include <hip/hip_runtime.h>

// ---------------------------------------------------------------------------
// Batched GEMM: out[b,h] = x1[b,h] (2048x128) @ x2[b,h] (128x2048), fp32.
// fp32 is decomposed into bf16 hi/lo (a = ah + al) once at LDS-fill time;
// products computed as Ah*Bh + Ah*Bl + Al*Bh with v_wmma_f32_16x16x32_bf16
// (fp32 accumulation) -> ~2^-16 relative error vs exact fp32 GEMM.
// ---------------------------------------------------------------------------

typedef __attribute__((ext_vector_type(4)))  float   v4f;
typedef __attribute__((ext_vector_type(8)))  float   v8f;
typedef __attribute__((ext_vector_type(4)))  __bf16  v4bf;
typedef __attribute__((ext_vector_type(8)))  __bf16  v8bf;
typedef __attribute__((ext_vector_type(16))) __bf16  v16bf;

#define SDIM 2048   // rows of x1 / out
#define TDIM 2048   // cols of x2 / out
#define KDIM 128    // contraction
#define NBATCH 32   // 2*16
#define TILE 128    // workgroup output tile: 128x128
#define LDSS 136    // bf16 row stride in LDS: 16B-aligned rows, bank-friendly
#define TILE_ELEMS (TILE * LDSS)

__device__ __forceinline__ void split_hl(float x, __bf16 &h, __bf16 &l) {
  h = (__bf16)x;                 // RNE round to top mantissa bits
  l = (__bf16)(x - (float)h);    // residual captures next ~8 mantissa bits
}

__global__ __launch_bounds__(256)
void EfficientMemoryGEMM_bf16split_kernel(const float* __restrict__ x1,
                                          const float* __restrict__ x2,
                                          float* __restrict__ out) {
  extern __shared__ __align__(16) __bf16 smem[];
  __bf16* Ah = smem;                       // [128][LDSS]  (m, k)
  __bf16* Al = Ah + TILE_ELEMS;            // [128][LDSS]  (m, k)
  __bf16* Bh = Al + TILE_ELEMS;            // [128][LDSS]  (n, k)  transposed
  __bf16* Bl = Bh + TILE_ELEMS;            // [128][LDSS]  (n, k)  transposed

  const int tid = threadIdx.x;
  const int bx  = blockIdx.x;              // N tile index (0..15)
  const int by  = blockIdx.y;              // M tile index (0..15)
  const int bz  = blockIdx.z;              // batch*head   (0..31)

  const float* Ag = x1  + (size_t)bz * SDIM * KDIM + (size_t)(by * TILE) * KDIM;
  const float* Bg = x2  + (size_t)bz * KDIM * TDIM + (size_t)(bx * TILE);
  float*       Cg = out + (size_t)bz * SDIM * TDIM + (size_t)(by * TILE) * TDIM
                        + (size_t)(bx * TILE);

  // ---- global -> LDS, splitting fp32 into bf16 hi/lo planes ---------------
  // A tile: 128 rows x 128 k (k contiguous). 4096 float4 slots / 256 threads.
  #pragma unroll
  for (int i = 0; i < 16; ++i) {
    const int idx = tid + i * 256;
    const int row = idx >> 5;              // 32 float4 per 128-elem row
    const int f4  = idx & 31;
    const v4f v = *(const v4f*)(Ag + row * KDIM + f4 * 4);
    v4bf h, l;
    #pragma unroll
    for (int c = 0; c < 4; ++c) {
      __bf16 hh, ll;
      split_hl(v[c], hh, ll);
      h[c] = hh;
      l[c] = ll;
    }
    *(v4bf*)(Ah + row * LDSS + f4 * 4) = h;   // ds_store_b64
    *(v4bf*)(Al + row * LDSS + f4 * 4) = l;
  }
  // B tile: global is [k][n] (n contiguous, coalesced); store transposed
  // [n][k] so each lane's WMMA fragment is two contiguous 8-element K runs.
  #pragma unroll
  for (int i = 0; i < 16; ++i) {
    const int idx = tid + i * 256;
    const int k   = idx >> 5;
    const int n   = (idx & 31) * 4;
    const v4f v = *(const v4f*)(Bg + (size_t)k * TDIM + n);
    #pragma unroll
    for (int c = 0; c < 4; ++c) {
      __bf16 h, l;
      split_hl(v[c], h, l);
      Bh[(n + c) * LDSS + k] = h;
      Bl[(n + c) * LDSS + k] = l;
    }
  }
  __syncthreads();

  // ---- per-wave compute: 128x16 output strip ------------------------------
  const int wave = tid >> 5;               // 0..7 -> 16-col strip
  const int lane = tid & 31;
  const int n0   = wave * 16;
  const int mn   = lane & 15;              // M (for A) / N (for B,C) index
  const int hi   = (lane >> 4) & 1;        // lane-half selects K sub-block

  // B fragments (32x16 per k-step) held in registers across all M subtiles.
  // ISA B layout (16-bit, K=32): lanes 0-15: elems 0-7 = K0..7,
  // elems 8-15 = K16..23; lanes 16-31: K8..15 / K24..31.
  v16bf bhf[4], blf[4];
  #pragma unroll
  for (int ks = 0; ks < 4; ++ks) {
    const int koff = ks * 32 + hi * 8;
    const __bf16* ph = Bh + (n0 + mn) * LDSS + koff;
    const __bf16* pl = Bl + (n0 + mn) * LDSS + koff;
    const v8bf h0 = *(const v8bf*)(ph);      // ds_load_b128
    const v8bf h1 = *(const v8bf*)(ph + 16);
    const v8bf l0 = *(const v8bf*)(pl);
    const v8bf l1 = *(const v8bf*)(pl + 16);
    #pragma unroll
    for (int j = 0; j < 8; ++j) {
      bhf[ks][j] = h0[j]; bhf[ks][8 + j] = h1[j];
      blf[ks][j] = l0[j]; blf[ks][8 + j] = l1[j];
    }
  }

  #pragma unroll
  for (int mt = 0; mt < 8; ++mt) {
    v8f acc = {};
    #pragma unroll
    for (int ks = 0; ks < 4; ++ks) {
      // ISA A layout (16-bit, 16x32): lanes 0-15: elems 0-7 = K0..7,
      // elems 8-15 = K16..23; lanes 16-31: K8..15 / K24..31.
      const __bf16* pah = Ah + (mt * 16 + mn) * LDSS + ks * 32 + hi * 8;
      const __bf16* pal = Al + (mt * 16 + mn) * LDSS + ks * 32 + hi * 8;
      const v8bf a0 = *(const v8bf*)(pah);
      const v8bf a1 = *(const v8bf*)(pah + 16);
      const v8bf c0 = *(const v8bf*)(pal);
      const v8bf c1 = *(const v8bf*)(pal + 16);
      v16bf af, lf;
      #pragma unroll
      for (int j = 0; j < 8; ++j) {
        af[j] = a0[j]; af[8 + j] = a1[j];
        lf[j] = c0[j]; lf[8 + j] = c1[j];
      }
      acc = __builtin_amdgcn_wmma_f32_16x16x32_bf16(false, af, false, bhf[ks],
                                                    (short)0, acc, false, false);
      acc = __builtin_amdgcn_wmma_f32_16x16x32_bf16(false, af, false, blf[ks],
                                                    (short)0, acc, false, false);
      acc = __builtin_amdgcn_wmma_f32_16x16x32_bf16(false, lf, false, bhf[ks],
                                                    (short)0, acc, false, false);
    }
    // C layout: VGPR r -> lanes 0-15: (M=r, N=lane); lanes 16-31: (M=r+8).
    const int row0 = mt * 16 + hi * 8;
    #pragma unroll
    for (int r = 0; r < 8; ++r) {
      Cg[(size_t)(row0 + r) * TDIM + (n0 + mn)] = acc[r];
    }
  }
}

extern "C" void kernel_launch(void* const* d_in, const int* in_sizes, int n_in,
                              void* d_out, int out_size, void* d_ws, size_t ws_size,
                              hipStream_t stream) {
  (void)in_sizes; (void)n_in; (void)d_ws; (void)ws_size; (void)out_size;
  const float* x1 = (const float*)d_in[0];
  const float* x2 = (const float*)d_in[1];
  // d_in[2] = rank: only affects discarded side-effect pipeline; unused.
  float* out = (float*)d_out;

  const size_t shmem = (size_t)4 * TILE_ELEMS * sizeof(__bf16);  // ~136 KB
  static_assert(4 * TILE_ELEMS * sizeof(__bf16) <= 320 * 1024, "LDS budget");
  // Allow >64KB dynamic LDS (no-op/harmless if already permitted).
  (void)hipFuncSetAttribute((const void*)EfficientMemoryGEMM_bf16split_kernel,
                            hipFuncAttributeMaxDynamicSharedMemorySize,
                            (int)shmem);

  dim3 grid(TDIM / TILE, SDIM / TILE, NBATCH);   // 16 x 16 x 32
  dim3 block(256);                                // 8 waves (wave32)
  EfficientMemoryGEMM_bf16split_kernel<<<grid, block, shmem, stream>>>(x1, x2, out);
}
